// LoRALinear_16716012716311
// MI455X (gfx1250) — matrix operand order repaired and checked by reading, not compiled, not run on hardware
//
#include <hip/hip_runtime.h>

// ---------------------------------------------------------------------------
// LoRA linear on MI455X (gfx1250, wave32).
//   out[M,N] = x[M,K] @ W_eff[N,K]^T + b,  W_eff = W + 2*(gB@gA + lB@lA)
// M=16384 (B*S), K=1024, N=4096, r=8.
//
// Pass 1a: fold both rank-8 adapters into W_eff (bf16 in ws, 8 MB).
// Pass 1b (if ws fits): convert x to bf16 once (32 MB in ws).
// Pass 2:  bf16 WMMA GEMM, fp32 accumulate.
//   - preferred: 128x256 block tile, 64x64 wave tile (16 wmma : 16 ds_b128),
//     double-buffered LDS fed by GLOBAL_LOAD_ASYNC_TO_LDS_B128 (ASYNCcnt).
//   - fallback (small ws): 128x128 tile, x converted fp32->bf16 in-kernel.
// ---------------------------------------------------------------------------

typedef __attribute__((ext_vector_type(16))) __bf16        v16bf;
typedef __attribute__((ext_vector_type(8)))  float         v8f;
typedef __attribute__((ext_vector_type(8)))  unsigned int  v8u;

#define D_IN   1024
#define D_OUT  4096
#define M_TOT  16384
#define TILE_K 32
#define LDS_STRIDE (TILE_K + 8)   // padded LDS row stride (ushort units)

__device__ __forceinline__ unsigned short f32_to_bf16_rne(float f) {
  unsigned int u = __float_as_uint(f);
  u += 0x7FFFu + ((u >> 16) & 1u);          // round-to-nearest-even
  return (unsigned short)(u >> 16);
}

// ---------------------------------------------------------------------------
// Pass 1a: W_eff[o,d] = W[o,d] + 2*(sum_r gB[o,r]*gA[r,d] + lB[o,r]*lA[r,d])
// ---------------------------------------------------------------------------
__global__ void build_weff_kernel(const float* __restrict__ W,
                                  const float* __restrict__ gA,
                                  const float* __restrict__ gB,
                                  const float* __restrict__ lA,
                                  const float* __restrict__ lB,
                                  unsigned short* __restrict__ weff) {
  int idx = blockIdx.x * blockDim.x + threadIdx.x;
  int o = idx >> 10;
  int d = idx & (D_IN - 1);
  float s = 0.0f;
#pragma unroll
  for (int r = 0; r < 8; ++r)
    s += gB[o * 8 + r] * gA[r * D_IN + d] + lB[o * 8 + r] * lA[r * D_IN + d];
  weff[idx] = f32_to_bf16_rne(W[idx] + 2.0f * s);
}

// ---------------------------------------------------------------------------
// Pass 1b: x (fp32) -> bf16, 4 elements per thread.
// ---------------------------------------------------------------------------
__global__ void convert_x_kernel(const float* __restrict__ X,
                                 unsigned short* __restrict__ Xb) {
  int v = blockIdx.x * blockDim.x + threadIdx.x;
  float4 f = *(const float4*)&X[(size_t)v * 4];
  unsigned int p0 = (unsigned int)f32_to_bf16_rne(f.x) |
                    ((unsigned int)f32_to_bf16_rne(f.y) << 16);
  unsigned int p1 = (unsigned int)f32_to_bf16_rne(f.z) |
                    ((unsigned int)f32_to_bf16_rne(f.w) << 16);
  *(uint2*)&Xb[(size_t)v * 4] = make_uint2(p0, p1);
}

// ---------------------------------------------------------------------------
// Fragment loaders (LDS layouts per CDNA5 ISA §7.12.2):
//   A: lane row M = lane%16, K chunks at 8*hh and 16+8*hh
//   B: lane col N = lane%16, K chunks at 8*hh and 16+8*hh
// ---------------------------------------------------------------------------
__device__ __forceinline__ v16bf load_afrag(const unsigned short* __restrict__ Al,
                                            int arow, int hh) {
  uint4 q0 = *(const uint4*)&Al[arow * LDS_STRIDE + 8 * hh];
  uint4 q1 = *(const uint4*)&Al[arow * LDS_STRIDE + 16 + 8 * hh];
  v8u u;
  u[0] = q0.x; u[1] = q0.y; u[2] = q0.z; u[3] = q0.w;
  u[4] = q1.x; u[5] = q1.y; u[6] = q1.z; u[7] = q1.w;
  return __builtin_bit_cast(v16bf, u);
}

__device__ __forceinline__ v16bf load_bfrag(const unsigned short* __restrict__ Bl,
                                            int nrow, int hh) {
  uint4 q0 = *(const uint4*)&Bl[nrow * LDS_STRIDE + 8 * hh];
  uint4 q1 = *(const uint4*)&Bl[nrow * LDS_STRIDE + 16 + 8 * hh];
  v8u u;
  u[0] = q0.x; u[1] = q0.y; u[2] = q0.z; u[3] = q0.w;
  u[4] = q1.x; u[5] = q1.y; u[6] = q1.z; u[7] = q1.w;
  return __builtin_bit_cast(v16bf, u);
}

__device__ __forceinline__ void wait_async0() {
  asm volatile("s_wait_asynccnt 0x0" ::: "memory");
}

// ---------------------------------------------------------------------------
// Async staging: copy ROWS x 32 bf16 tile (gstride D_IN) into LDS with
// GLOBAL_LOAD_ASYNC_TO_LDS_B128 (16 bytes per lane).
// ---------------------------------------------------------------------------
template <int ROWS>
__device__ __forceinline__ void stage_async_tile(const unsigned short* __restrict__ g0,
                                                 unsigned short* lds, int t) {
#pragma unroll
  for (int i = 0; i < ROWS / 64; ++i) {   // 256 threads x 16B cover 64 rows
    int v   = t + 256 * i;
    int row = v >> 2;                     // 4 chunks per row
    int c   = v & 3;
    unsigned long long ga = (unsigned long long)(size_t)(g0 + (size_t)row * D_IN + c * 8);
    unsigned int       la = (unsigned int)(size_t)(&lds[row * LDS_STRIDE + c * 8]);
    asm volatile("global_load_async_to_lds_b128 %0, %1, off"
                 :: "v"(la), "v"(ga) : "memory");
  }
}

// ---------------------------------------------------------------------------
// Pass 2 (preferred): 128(M) x 256(N) block tile, 8 waves as 2(M) x 4(N),
// each wave owns 64x64 = 4x4 WMMA tiles. Async double-buffered LDS.
// ---------------------------------------------------------------------------
#define BT_M 128
#define BT_N 256

__global__ __launch_bounds__(256)
void lora_gemm_bf16_kernel(const unsigned short* __restrict__ Xb,
                           const unsigned short* __restrict__ Wf,
                           const float* __restrict__ bias,
                           float* __restrict__ out) {
  __shared__ alignas(16) unsigned short Alds[2][BT_M * LDS_STRIDE];
  __shared__ alignas(16) unsigned short Blds[2][BT_N * LDS_STRIDE];

  const int t  = threadIdx.x;
  const int n0 = blockIdx.x * BT_N;
  const int m0 = blockIdx.y * BT_M;

  const int lane  = t & 31;
  const int wave  = t >> 5;
  const int waveM = (wave >> 2) * 64;   // 0 or 64
  const int waveN = (wave & 3) * 64;    // 0,64,128,192
  const int lrow  = lane & 15;
  const int hh    = lane >> 4;

  const unsigned short* Ag = Xb + (size_t)m0 * D_IN;
  const unsigned short* Bg = Wf + (size_t)n0 * D_IN;

  v8f acc[4][4] = {};

  // prologue: stage K-tile 0 into buffer 0
  stage_async_tile<BT_M>(Ag, Alds[0], t);
  stage_async_tile<BT_N>(Bg, Blds[0], t);
  wait_async0();
  __syncthreads();

  int cur = 0;
  for (int k0 = 0; k0 < D_IN; k0 += TILE_K) {
    int nxt = cur ^ 1;
    if (k0 + TILE_K < D_IN) {           // stage next tile while computing this one
      stage_async_tile<BT_M>(Ag + k0 + TILE_K, Alds[nxt], t);
      stage_async_tile<BT_N>(Bg + k0 + TILE_K, Blds[nxt], t);
    }

    // ---- compute: 4 B frags reused against 4 A frags -> 16 WMMAs ----
    v16bf bfrag[4];
#pragma unroll
    for (int n = 0; n < 4; ++n)
      bfrag[n] = load_bfrag(Blds[cur], waveN + n * 16 + lrow, hh);
#pragma unroll
    for (int m = 0; m < 4; ++m) {
      v16bf afrag = load_afrag(Alds[cur], waveM + m * 16 + lrow, hh);
#pragma unroll
      for (int n = 0; n < 4; ++n)
        acc[m][n] = __builtin_amdgcn_wmma_f32_16x16x32_bf16(
            false, afrag, false, bfrag[n], (short)0, acc[m][n], false, false);
    }

    wait_async0();                      // my async writes to LDS landed
    __syncthreads();                    // everyone done reading cur / writing nxt
    cur = nxt;
  }

  // ---- epilogue: D layout VGPR v -> M = v + 8*hh, lane%16 -> N ----
#pragma unroll
  for (int n = 0; n < 4; ++n) {
    int col  = n0 + waveN + n * 16 + lrow;
    float bb = bias[col];
#pragma unroll
    for (int m = 0; m < 4; ++m) {
      int rbase = m0 + waveM + m * 16 + 8 * hh;
#pragma unroll
      for (int v = 0; v < 8; ++v)
        out[(size_t)(rbase + v) * D_OUT + col] = acc[m][n][v] + bb;
    }
  }
}

// ---------------------------------------------------------------------------
// Pass 2 (fallback, small ws): 128x128 tile, x converted fp32->bf16 in-kernel.
// ---------------------------------------------------------------------------
#define FT_M 128
#define FT_N 128

__global__ __launch_bounds__(256)
void lora_gemm_f32x_kernel(const float* __restrict__ X,
                           const unsigned short* __restrict__ Wf,
                           const float* __restrict__ bias,
                           float* __restrict__ out) {
  __shared__ alignas(16) unsigned short Alds[FT_M * LDS_STRIDE];
  __shared__ alignas(16) unsigned short Blds[FT_N * LDS_STRIDE];

  const int t  = threadIdx.x;
  const int n0 = blockIdx.x * FT_N;
  const int m0 = blockIdx.y * FT_M;

  const int lane  = t & 31;
  const int wave  = t >> 5;
  const int waveM = (wave >> 2) * 64;
  const int waveN = (wave & 3) * 32;
  const int lrow  = lane & 15;
  const int hh    = lane >> 4;

  v8f acc[4][2] = {};

  for (int k0 = 0; k0 < D_IN; k0 += TILE_K) {
#pragma unroll
    for (int i = 0; i < 4; ++i) {
      int v   = t + 256 * i;
      int row = v >> 3;
      int c4  = v & 7;
      float4 f = *(const float4*)&X[(size_t)(m0 + row) * D_IN + k0 + c4 * 4];
      unsigned int p0 = (unsigned int)f32_to_bf16_rne(f.x) |
                        ((unsigned int)f32_to_bf16_rne(f.y) << 16);
      unsigned int p1 = (unsigned int)f32_to_bf16_rne(f.z) |
                        ((unsigned int)f32_to_bf16_rne(f.w) << 16);
      *(uint2*)&Alds[row * LDS_STRIDE + c4 * 4] = make_uint2(p0, p1);
    }
#pragma unroll
    for (int i = 0; i < 2; ++i) {
      int v   = t + 256 * i;
      int row = v >> 2;
      int c   = v & 3;
      uint4 q = *(const uint4*)&Wf[(size_t)(n0 + row) * D_IN + k0 + c * 8];
      *(uint4*)&Blds[row * LDS_STRIDE + c * 8] = q;
    }
    if (k0 + TILE_K < D_IN) {
      __builtin_prefetch(&X[(size_t)(m0 + (t >> 1)) * D_IN + k0 + TILE_K], 0, 0);
      __builtin_prefetch(&Wf[(size_t)(n0 + (t >> 1)) * D_IN + k0 + TILE_K], 0, 0);
    }
    __syncthreads();

    v16bf bfrag[2];
#pragma unroll
    for (int n = 0; n < 2; ++n)
      bfrag[n] = load_bfrag(Blds, waveN + n * 16 + lrow, hh);
#pragma unroll
    for (int m = 0; m < 4; ++m) {
      v16bf afrag = load_afrag(Alds, waveM + m * 16 + lrow, hh);
#pragma unroll
      for (int n = 0; n < 2; ++n)
        acc[m][n] = __builtin_amdgcn_wmma_f32_16x16x32_bf16(
            false, afrag, false, bfrag[n], (short)0, acc[m][n], false, false);
    }
    __syncthreads();
  }

#pragma unroll
  for (int n = 0; n < 2; ++n) {
    int col  = n0 + waveN + n * 16 + lrow;
    float bb = bias[col];
#pragma unroll
    for (int m = 0; m < 4; ++m) {
      int rbase = m0 + waveM + m * 16 + 8 * hh;
#pragma unroll
      for (int v = 0; v < 8; ++v)
        out[(size_t)(rbase + v) * D_OUT + col] = acc[m][n][v] + bb;
    }
  }
}

// ---------------------------------------------------------------------------
extern "C" void kernel_launch(void* const* d_in, const int* in_sizes, int n_in,
                              void* d_out, int out_size, void* d_ws, size_t ws_size,
                              hipStream_t stream) {
  const float* x  = (const float*)d_in[0];   // [8,2048,1024]
  const float* W  = (const float*)d_in[1];   // [4096,1024]
  const float* b  = (const float*)d_in[2];   // [4096]
  const float* gA = (const float*)d_in[3];   // [8,1024]
  const float* gB = (const float*)d_in[4];   // [4096,8]
  const float* lA = (const float*)d_in[5];   // [8,1024]
  const float* lB = (const float*)d_in[6];   // [4096,8]
  float* out = (float*)d_out;                // [8,2048,4096]

  const size_t weff_bytes = (size_t)D_OUT * D_IN * 2;   // 8 MB
  const size_t xbf_bytes  = (size_t)M_TOT * D_IN * 2;   // 32 MB

  unsigned short* weff = (unsigned short*)d_ws;

  build_weff_kernel<<<(D_OUT * D_IN) / 256, 256, 0, stream>>>(W, gA, gB, lA, lB, weff);

  if (ws_size >= weff_bytes + xbf_bytes) {
    unsigned short* xbf = (unsigned short*)((char*)d_ws + weff_bytes);
    convert_x_kernel<<<(M_TOT * D_IN / 4) / 256, 256, 0, stream>>>(x, xbf);
    dim3 grid(D_OUT / BT_N, M_TOT / BT_M);              // (16, 128)
    lora_gemm_bf16_kernel<<<grid, 256, 0, stream>>>(xbf, weff, b, out);
  } else {
    dim3 grid(D_OUT / FT_N, M_TOT / FT_M);              // (32, 128)
    lora_gemm_f32x_kernel<<<grid, 256, 0, stream>>>(x, weff, b, out);
  }
}
